// TurboQuantRuntime_73297911873566
// MI455X (gfx1250) — matrix operand order repaired and checked by reading, not compiled, not run on hardware
//
#include <hip/hip_runtime.h>

typedef __attribute__((ext_vector_type(16))) _Float16 v16h;
typedef __attribute__((ext_vector_type(8)))  float    v8f;

#define HEADS 32
#define SQN   512
#define SKV   4096
#define DH    128

__device__ __forceinline__ v8f wmma16(v16h a, v16h b, v8f c) {
  return __builtin_amdgcn_wmma_f32_16x16x32_f16(false, a, false, b, (short)0, c,
                                                false, false);
}

// Async global->LDS copy of 16 bytes per lane (ASYNCcnt-tracked, CDNA5).
__device__ __forceinline__ void async_b128(unsigned lds_off, const _Float16* g) {
  asm volatile("global_load_async_to_lds_b128 %0, %1, off"
               :: "v"(lds_off), "v"(g) : "memory");
}
__device__ __forceinline__ void wait_async_le12() {
  asm volatile("s_wait_asynccnt 0xc" ::: "memory");
}
__device__ __forceinline__ void wait_async_0() {
  asm volatile("s_wait_asynccnt 0x0" ::: "memory");
}

// A tile (16 x 32) from row-major src (leading dim ld), k-window [kbase, kbase+32)
__device__ __forceinline__ v16h load_a(const _Float16* src, int ld, int kbase, int lane) {
  int m = lane & 15, g = lane >> 4;
  const _Float16* p = src + (size_t)m * ld + kbase + g * 16;
  v16h o;
#pragma unroll
  for (int v = 0; v < 8; ++v) {
    int off = 2 * v;
    o[2 * v]     = p[off];
    o[2 * v + 1] = p[off + 1];
  }
  return o;
}

// B tile (32 x 16): B(k,n) = src[(nbase+n)*ld + koff + k]  (B = W^T, W row-major)
__device__ __forceinline__ v16h load_bt(const _Float16* src, int ld, int nbase,
                                        int koff, int lane) {
  int n = lane & 15, g = lane >> 4;
  const _Float16* p = src + (size_t)(nbase + n) * ld + koff + g * 16;
  v16h o;
#pragma unroll
  for (int v = 0; v < 8; ++v) { o[2 * v] = p[2 * v]; o[2 * v + 1] = p[2 * v + 1]; }
  return o;
}

// B tile (32 x 16): B(k,n) = src[k*ld + nbase + n]  (row-major K x N slab, e.g. V in LDS)
__device__ __forceinline__ v16h load_bs(const _Float16* src, int ld, int nbase, int lane) {
  int n = lane & 15, g = lane >> 4;
  v16h o;
#pragma unroll
  for (int v = 0; v < 8; ++v) {
    int k0 = g * 16 + 2 * v;
    o[2 * v]     = src[(size_t)k0 * ld + nbase + n];
    o[2 * v + 1] = src[(size_t)(k0 + 1) * ld + nbase + n];
  }
  return o;
}

// ---------------------------------------------------------------- conversions
__global__ void convert_kernel(const float* __restrict__ KR, const float* __restrict__ QJ,
                               const float* __restrict__ VR,
                               _Float16* __restrict__ Rh,  _Float16* __restrict__ Rth,
                               _Float16* __restrict__ QJh, _Float16* __restrict__ Vrh,
                               _Float16* __restrict__ Vrth) {
  int i = blockIdx.x, d = threadIdx.x;
  float r = KR[i * 128 + d];
  Rh[i * 128 + d]  = (_Float16)r;
  Rth[d * 128 + i] = (_Float16)r;
  QJh[i * 128 + d] = (_Float16)QJ[i * 128 + d];
  float w = VR[i * 128 + d];
  Vrh[i * 128 + d]  = (_Float16)w;
  Vrth[d * 128 + i] = (_Float16)w;
}

// ---------------------------------------------------------------- key compress
__global__ void key_compress_kernel(const float* __restrict__ K,
                                    const _Float16* __restrict__ Rh,
                                    const _Float16* __restrict__ Rth,
                                    const _Float16* __restrict__ QJh,
                                    const float* __restrict__ cent,
                                    const float* __restrict__ bnd,
                                    _Float16* __restrict__ Kext) {
  __shared__ float    kbuf[32 * 128];
  __shared__ _Float16 h0[32 * 128];
  __shared__ _Float16 h1[32 * 128];
  __shared__ float    norms[32];
  __shared__ float    sb[7], sc[8];

  const int tid = threadIdx.x;
  const int head = blockIdx.y;
  const int rowbase = blockIdx.x * 32;
  if (tid < 8) sc[tid] = cent[tid];
  if (tid < 7) sb[tid] = bnd[tid];

  // load rows + norms: thread pair per row, vectorized b128
  const int row = tid >> 1, half = tid & 1;
  const float4* kr4 = (const float4*)(K + ((size_t)head * SKV + rowbase + row) * DH + half * 64);
  float4* kb4 = (float4*)&kbuf[row * 128 + half * 64];
  float ss = 0.f;
#pragma unroll
  for (int j = 0; j < 16; ++j) {
    float4 x = kr4[j];
    kb4[j] = x;
    ss += x.x * x.x + x.y * x.y + x.z * x.z + x.w * x.w;
  }
  ss += __shfl_xor(ss, 1, 32);
  float nrm = sqrtf(ss);
  if (half == 0) norms[row] = nrm;
  float inv = 1.f / (nrm + 1e-8f);
#pragma unroll 4
  for (int j = 0; j < 64; ++j)
    h0[row * 128 + half * 64 + j] = (_Float16)(kbuf[row * 128 + half * 64 + j] * inv);
  __syncthreads();

  const int w = tid >> 5, lane = tid & 31;
  const int g = lane >> 4, n = lane & 15;
  const _Float16* A0 = h0 + w * 16 * 128;
  _Float16* KeH = Kext + ((size_t)head * SKV + rowbase) * 256;

  // stage 1: rotated = kn @ R^T, quantize -> h1
  for (int it = 0; it < 8; ++it) {
    v8f acc = {};
#pragma unroll
    for (int kc = 0; kc < 4; ++kc)
      acc = wmma16(load_a(A0, 128, kc * 32, lane),
                   load_bt(Rh, 128, it * 16, kc * 32, lane), acc);
#pragma unroll
    for (int r = 0; r < 8; ++r) {
      float x = acc[r];
      int idx = 0;
#pragma unroll
      for (int j = 0; j < 7; ++j) idx += (x > sb[j]) ? 1 : 0;
      h1[(w * 16 + r + 8 * g) * 128 + it * 16 + n] = (_Float16)sc[idx];
    }
  }
  __syncthreads();

  // stage 2: mse = c @ R ; residual ; res-norm ; Kext[0:128]
  const _Float16* A1 = h1 + w * 16 * 128;
  float rs[8];
#pragma unroll
  for (int r = 0; r < 8; ++r) rs[r] = 0.f;
  for (int dt = 0; dt < 8; ++dt) {
    v8f acc = {};
#pragma unroll
    for (int kc = 0; kc < 4; ++kc)
      acc = wmma16(load_a(A1, 128, kc * 32, lane),
                   load_bt(Rth, 128, dt * 16, kc * 32, lane), acc);
#pragma unroll
    for (int r = 0; r < 8; ++r) {
      int grow = w * 16 + r + 8 * g;
      float nf = norms[grow];
      float nq = (float)(_Float16)nf;          // stored-f16 norm, like reference
      float mseu = acc[r];
      float res = kbuf[grow * 128 + dt * 16 + n] - mseu * nf;
      rs[r] += res * res;
      h0[grow * 128 + dt * 16 + n] = (_Float16)res;  // reuse h0 (own-wave slice)
      KeH[(size_t)grow * 256 + dt * 16 + n] = (_Float16)(mseu * nq);
    }
  }
  float rq[8];
#pragma unroll
  for (int r = 0; r < 8; ++r) {
#pragma unroll
    for (int o = 8; o >= 1; o >>= 1) rs[r] += __shfl_xor(rs[r], o, 16);
    rq[r] = (float)(_Float16)sqrtf(rs[r]);     // res_norm rounded to f16
  }
  __syncthreads();

  // stage 3: signs = sign(res @ qjl^T); fold correction*res_norm into Kext[128:256]
  const float corr = 0.009791516697777346f;    // sqrt(pi/2)/128
  for (int jt = 0; jt < 8; ++jt) {
    v8f acc = {};
#pragma unroll
    for (int kc = 0; kc < 4; ++kc)
      acc = wmma16(load_a(A0, 128, kc * 32, lane),
                   load_bt(QJh, 128, jt * 16, kc * 32, lane), acc);
#pragma unroll
    for (int r = 0; r < 8; ++r) {
      int grow = w * 16 + r + 8 * g;
      float s = (acc[r] >= 0.f) ? 1.f : -1.f;
      KeH[(size_t)grow * 256 + 128 + jt * 16 + n] = (_Float16)(s * corr * rq[r]);
    }
  }
}

// ---------------------------------------------------------------- value compress
__global__ void val_compress_kernel(const float* __restrict__ V,
                                    const _Float16* __restrict__ Vrh,
                                    const _Float16* __restrict__ Vrth,
                                    const float* __restrict__ cent,
                                    const float* __restrict__ bnd,
                                    _Float16* __restrict__ Va) {
  __shared__ _Float16 h0[32 * 128];
  __shared__ _Float16 h1[32 * 128];
  __shared__ float    norms[32];
  __shared__ float    sb[15], sc[16];

  const int tid = threadIdx.x;
  const int head = blockIdx.y;
  const int rowbase = blockIdx.x * 32;
  if (tid < 16) sc[tid] = cent[tid];
  if (tid < 15) sb[tid] = bnd[tid];

  const int row = tid >> 1, half = tid & 1;
  const float4* vr4 = (const float4*)(V + ((size_t)head * SKV + rowbase + row) * DH + half * 64);
  float4 vreg[16];
  float ss = 0.f;
#pragma unroll
  for (int j = 0; j < 16; ++j) {
    float4 x = vr4[j];
    vreg[j] = x;
    ss += x.x * x.x + x.y * x.y + x.z * x.z + x.w * x.w;
  }
  ss += __shfl_xor(ss, 1, 32);
  float nrm = sqrtf(ss);
  if (half == 0) norms[row] = nrm;
  float inv = 1.f / (nrm + 1e-8f);
#pragma unroll
  for (int j = 0; j < 16; ++j) {
    h0[row * 128 + half * 64 + 4 * j + 0] = (_Float16)(vreg[j].x * inv);
    h0[row * 128 + half * 64 + 4 * j + 1] = (_Float16)(vreg[j].y * inv);
    h0[row * 128 + half * 64 + 4 * j + 2] = (_Float16)(vreg[j].z * inv);
    h0[row * 128 + half * 64 + 4 * j + 3] = (_Float16)(vreg[j].w * inv);
  }
  __syncthreads();

  const int w = tid >> 5, lane = tid & 31;
  const int g = lane >> 4, n = lane & 15;
  const _Float16* A0 = h0 + w * 16 * 128;
  _Float16* VaH = Va + ((size_t)head * SKV + rowbase) * 128;

  for (int it = 0; it < 8; ++it) {
    v8f acc = {};
#pragma unroll
    for (int kc = 0; kc < 4; ++kc)
      acc = wmma16(load_a(A0, 128, kc * 32, lane),
                   load_bt(Vrh, 128, it * 16, kc * 32, lane), acc);
#pragma unroll
    for (int r = 0; r < 8; ++r) {
      float x = acc[r];
      int idx = 0;
#pragma unroll
      for (int j = 0; j < 15; ++j) idx += (x > sb[j]) ? 1 : 0;
      h1[(w * 16 + r + 8 * g) * 128 + it * 16 + n] = (_Float16)sc[idx];
    }
  }
  __syncthreads();

  const _Float16* A1 = h1 + w * 16 * 128;
  for (int dt = 0; dt < 8; ++dt) {
    v8f acc = {};
#pragma unroll
    for (int kc = 0; kc < 4; ++kc)
      acc = wmma16(load_a(A1, 128, kc * 32, lane),
                   load_bt(Vrth, 128, dt * 16, kc * 32, lane), acc);
#pragma unroll
    for (int r = 0; r < 8; ++r) {
      int grow = w * 16 + r + 8 * g;
      float nq = (float)(_Float16)norms[grow];
      VaH[(size_t)grow * 128 + dt * 16 + n] = (_Float16)(acc[r] * nq);
    }
  }
}

// ---------------------------------------------------------------- Q_ext build
__global__ void q_build_kernel(const float* __restrict__ Q,
                               const _Float16* __restrict__ QJh,
                               _Float16* __restrict__ Qe) {
  __shared__ _Float16 h0[32 * 128];
  const int tid = threadIdx.x;
  const int head = blockIdx.y;
  const int rowbase = blockIdx.x * 32;
  const int row = tid >> 1, half = tid & 1;
  const float* qr = Q + ((size_t)head * SQN + rowbase + row) * DH + half * 64;
  _Float16* QeH = Qe + ((size_t)head * SQN + rowbase) * 256;
#pragma unroll 4
  for (int j = 0; j < 64; ++j) {
    _Float16 hv = (_Float16)qr[j];
    h0[row * 128 + half * 64 + j] = hv;
    QeH[(size_t)row * 256 + half * 64 + j] = hv;
  }
  __syncthreads();

  const int w = tid >> 5, lane = tid & 31;
  const int g = lane >> 4, n = lane & 15;
  const _Float16* A0 = h0 + w * 16 * 128;
  for (int jt = 0; jt < 8; ++jt) {
    v8f acc = {};
#pragma unroll
    for (int kc = 0; kc < 4; ++kc)
      acc = wmma16(load_a(A0, 128, kc * 32, lane),
                   load_bt(QJh, 128, jt * 16, kc * 32, lane), acc);
#pragma unroll
    for (int r = 0; r < 8; ++r) {
      int grow = w * 16 + r + 8 * g;
      QeH[(size_t)grow * 256 + 128 + jt * 16 + n] = (_Float16)acc[r];
    }
  }
}

// ---------------------------------------------------------------- attention
// Double-buffered ASYNC global->LDS staging of K_ext (16KB) + V (8KB) tiles,
// overlapped with WMMA on the live tile. 3 barriers/iter, ASYNCcnt pipeline.
__global__ void attn_kernel(const _Float16* __restrict__ Qe,
                            const _Float16* __restrict__ Ke,
                            const _Float16* __restrict__ Va,
                            float* __restrict__ Out) {
  __shared__ _Float16 Kl[2][32 * 256];
  __shared__ _Float16 Vl[2][32 * 128];
  __shared__ _Float16 Pl[4][16 * 32];

  const int tid = threadIdx.x, w = tid >> 5, lane = tid & 31;
  const int g = lane >> 4, n = lane & 15;
  const int head = blockIdx.y, qbase = blockIdx.x * 64;
  const float scal = 0.08838834764831845f;  // 1/sqrt(128)

  const _Float16* Qh = Qe + ((size_t)head * SQN + qbase + w * 16) * 256;
  const _Float16* Kh = Ke + (size_t)head * SKV * 256;
  const _Float16* Vh = Va + (size_t)head * SKV * 128;

  // issue 12 async b128 loads per thread: 8 for the K tile, 4 for the V tile
  auto issue_tile = [&](int buf, int kv) {
    const _Float16* kg = Kh + (size_t)kv * 256;
    const _Float16* vg = Vh + (size_t)kv * 128;
#pragma unroll
    for (int c = 0; c < 8; ++c) {
      int idx = c * 128 + tid;  // 16-byte units within 16KB tile
      async_b128((unsigned)(size_t)&Kl[buf][idx * 8], kg + idx * 8);
    }
#pragma unroll
    for (int c = 0; c < 4; ++c) {
      int idx = c * 128 + tid;  // 16-byte units within 8KB tile
      async_b128((unsigned)(size_t)&Vl[buf][idx * 8], vg + idx * 8);
    }
  };

  v16h qf[8];
#pragma unroll
  for (int kc = 0; kc < 8; ++kc) qf[kc] = load_a(Qh, 256, kc * 32, lane);

  v8f o[8];
  v8f zero = {};
#pragma unroll
  for (int dt = 0; dt < 8; ++dt) o[dt] = zero;
  float mrow[8], lrow[8];
#pragma unroll
  for (int r = 0; r < 8; ++r) { mrow[r] = -1e30f; lrow[r] = 0.f; }

  issue_tile(0, 0);
  const int NIT = SKV / 32;
  for (int it = 0; it < NIT; ++it) {
    const int buf = it & 1;
    __syncthreads();  // everyone done reading the buffer we are about to overwrite
    if (it + 1 < NIT) {
      issue_tile(buf ^ 1, (it + 1) * 32);
      wait_async_le12();          // previous 12-op batch (this tile) has landed
    } else {
      wait_async_0();
    }
    __syncthreads();  // tile `buf` visible to all waves

    v8f s0 = zero, s1 = zero;
#pragma unroll
    for (int kc = 0; kc < 8; ++kc) {
      s0 = wmma16(qf[kc], load_bt(&Kl[buf][0], 256, 0, kc * 32, lane), s0);
      s1 = wmma16(qf[kc], load_bt(&Kl[buf][0], 256, 16, kc * 32, lane), s1);
    }

#pragma unroll
    for (int r = 0; r < 8; ++r) {
      float x0 = s0[r] * scal, x1 = s1[r] * scal;
      float mx = fmaxf(x0, x1);
#pragma unroll
      for (int o2 = 8; o2 >= 1; o2 >>= 1) mx = fmaxf(mx, __shfl_xor(mx, o2, 16));
      float mnew = fmaxf(mrow[r], mx);
      float alpha = __expf(mrow[r] - mnew);
      float p0 = __expf(x0 - mnew), p1 = __expf(x1 - mnew);
      float sum = p0 + p1;
#pragma unroll
      for (int o2 = 8; o2 >= 1; o2 >>= 1) sum += __shfl_xor(sum, o2, 16);
      lrow[r] = lrow[r] * alpha + sum;
      mrow[r] = mnew;
#pragma unroll
      for (int dt = 0; dt < 8; ++dt) o[dt][r] *= alpha;
      int m = r + 8 * g;
      Pl[w][m * 32 + n]      = (_Float16)p0;
      Pl[w][m * 32 + 16 + n] = (_Float16)p1;
    }
    __syncthreads();  // P tiles visible

    v16h ap = load_a(&Pl[w][0], 32, 0, lane);
#pragma unroll
    for (int dt = 0; dt < 8; ++dt)
      o[dt] = wmma16(ap, load_bs(&Vl[buf][0], 128, dt * 16, lane), o[dt]);
  }

  float* Oh = Out + ((size_t)head * SQN + qbase + w * 16) * 128;
#pragma unroll
  for (int dt = 0; dt < 8; ++dt)
#pragma unroll
    for (int r = 0; r < 8; ++r) {
      int m = r + 8 * g;
      Oh[(size_t)m * 128 + dt * 16 + n] = o[dt][r] / lrow[r];
    }
}

// ---------------------------------------------------------------- launcher
extern "C" void kernel_launch(void* const* d_in, const int* in_sizes, int n_in,
                              void* d_out, int out_size, void* d_ws, size_t ws_size,
                              hipStream_t stream) {
  (void)in_sizes; (void)n_in; (void)out_size; (void)ws_size;
  const float* q  = (const float*)d_in[0];
  const float* k  = (const float*)d_in[1];
  const float* v  = (const float*)d_in[2];
  const float* KR = (const float*)d_in[3];
  const float* QJ = (const float*)d_in[4];
  const float* kc = (const float*)d_in[5];
  const float* kb = (const float*)d_in[6];
  const float* VR = (const float*)d_in[7];
  const float* vc = (const float*)d_in[8];
  const float* vb = (const float*)d_in[9];
  float* out = (float*)d_out;
  _Float16* ws = (_Float16*)d_ws;

  const size_t oRh  = 0;
  const size_t oRth = oRh + 128 * 128;
  const size_t oQJ  = oRth + 128 * 128;
  const size_t oVr  = oQJ + 128 * 128;
  const size_t oVrt = oVr + 128 * 128;
  const size_t oQe  = oVrt + 128 * 128;                       // H*SQ*256 halves
  const size_t oKe  = oQe + (size_t)HEADS * SQN * 256;        // H*SKV*256 halves
  const size_t oVa  = oKe + (size_t)HEADS * SKV * 256;        // H*SKV*128 halves

  convert_kernel<<<dim3(128), 128, 0, stream>>>(KR, QJ, VR, ws + oRh, ws + oRth,
                                                ws + oQJ, ws + oVr, ws + oVrt);
  key_compress_kernel<<<dim3(SKV / 32, HEADS), 64, 0, stream>>>(
      k, ws + oRh, ws + oRth, ws + oQJ, kc, kb, ws + oKe);
  val_compress_kernel<<<dim3(SKV / 32, HEADS), 64, 0, stream>>>(
      v, ws + oVr, ws + oVrt, vc, vb, ws + oVa);
  q_build_kernel<<<dim3(SQN / 32, HEADS), 64, 0, stream>>>(q, ws + oQJ, ws + oQe);
  attn_kernel<<<dim3(SQN / 64, HEADS), 128, 0, stream>>>(ws + oQe, ws + oKe,
                                                         ws + oVa, out);
}
